// RSSM_88321707475537
// MI455X (gfx1250) — matrix operand order repaired and checked by reading, not compiled, not run on hardware
//
#include <hip/hip_runtime.h>
#include <math.h>

// ---------------------------------------------------------------------------
// Types for CDNA5 WMMA (wave32, v_wmma_f32_16x16x32_bf16)
// ---------------------------------------------------------------------------
typedef unsigned short u16;
typedef unsigned long long uptr;
typedef __bf16  bfv16 __attribute__((ext_vector_type(16)));
typedef float   f32x8 __attribute__((ext_vector_type(8)));
typedef u16     u16x8 __attribute__((ext_vector_type(8)));

union FragU {            // 32 bytes: one 16-element bf16 WMMA operand
  bfv16 bf;
  u16x8 h[2];
};

// Pointer types expected by the async-to-LDS builtin (4 x i32 vector pointees)
typedef int v4i __attribute__((vector_size(16)));
typedef __attribute__((address_space(1))) v4i g_v4i;   // global (64-bit ptr)
typedef __attribute__((address_space(3))) v4i l_v4i;   // LDS    (32-bit ptr)

// Async global->LDS 16B copy per lane (GLOBAL_LOAD_ASYNC_TO_LDS_B128, ASYNCcnt)
__device__ __forceinline__ void async_b128(const void* g, void* l) {
  __builtin_amdgcn_global_load_async_to_lds_b128(
      (g_v4i*)(uptr)g, (l_v4i*)(unsigned int)(uptr)l, 0, 0);
}
__device__ __forceinline__ void wait_async0() {
  asm volatile("s_wait_asynccnt 0x0" ::: "memory");
}

__device__ __forceinline__ u16 f2bfu(float x) {  // f32 -> bf16 (RNE)
  unsigned int u = __float_as_uint(x);
  unsigned int r = u + 0x7FFFu + ((u >> 16) & 1u);
  return (u16)(r >> 16);
}

// ---------------------------------------------------------------------------
// Small utility kernels
// ---------------------------------------------------------------------------
__global__ void zero_f32_kernel(float* p, int n) {
  int i = blockIdx.x * 256 + threadIdx.x;
  if (i < n) p[i] = 0.f;
}

__global__ void cvt_f32_bf16_kernel(const float* __restrict__ in, u16* __restrict__ out, int n) {
  int i = blockIdx.x * 256 + threadIdx.x;
  if (i < n) out[i] = f2bfu(in[i]);
}

// W [N][K] f32 -> bf16 tiles of 64(n) x 32(k), [ni][kk], block index (nb, kt).
// Zero-padded; tile image == the GEMM kernel's LDS B layout (pure async copy).
__global__ void pack_wTtile_kernel(const float* __restrict__ W, u16* __restrict__ out,
                                   int N, int K, int NK) {
  int i = blockIdx.x * 256 + threadIdx.x;
  int total = ((N + 63) >> 6) * NK * 2048;
  if (i >= total) return;
  int blk = i >> 11, off = i & 2047;
  int nb = blk / NK, kt = blk - nb * NK;
  int ni = off >> 5, kk = off & 31;
  int n = nb * 64 + ni, k = kt * 32 + kk;
  u16 v = 0;
  if (n < N && k < K) v = f2bfu(W[n * K + k]);
  out[i] = v;
}

// ---------------------------------------------------------------------------
// WMMA GEMM (CONV=1: implicit-im2col A from NCHW bf16 tensor, k4/s2/p1;
//            CONV=0: plain row-major bf16 A with leading dim lda)
// Block: 256 threads = 8 wave32 waves as 4(M) x 2(N); block tile 128 x 64;
// each wave does a 32x32 tile via 2x2 v_wmma_f32_16x16x32_bf16.
// B tiles are pre-packed to the LDS image and staged with async-to-LDS b128.
// Conv path: OH/OW are powers of two and M % 128 == 0 -> shift/mask im2col.
// All tensors < 2^31 elements -> 32-bit index arithmetic throughout.
// ---------------------------------------------------------------------------
template <int CONV>
__global__ __launch_bounds__(256) void gemm_wmma_kernel(
    const u16* __restrict__ A, const u16* __restrict__ Bp,
    const float* __restrict__ bias, float* __restrict__ C,
    int M, int N, int K, int lda, int ldc,
    int Ci, int IH, int IW, int OW) {
  __shared__ __align__(16) u16 sA[128 * 32];  // A tile [mi][kk]
  __shared__ __align__(16) u16 sB[64 * 32];   // B tile [ni][kk] (async-staged)

  const int tid  = threadIdx.x;
  const int lane = tid & 31;
  const int wid  = tid >> 5;
  const int wm   = (wid >> 1) * 32;   // wave M offset in block tile
  const int wn   = (wid & 1) * 32;    // wave N offset in block tile
  const int m0   = blockIdx.x * 128;

  const int lOW = 31 - __builtin_clz((unsigned)OW);  // log2(OW) (conv only)
  const int lHW = 2 * lOW;                           // OH == OW for all layers

  f32x8 acc[2][2] = {};

  const int nk = (K + 31) >> 5;
  for (int kt = 0; kt < nk; ++kt) {
    const int k0 = kt << 5;

    // ---- issue async B-tile copy (4 KB contiguous, 16B per lane) ----
    {
      const u16* btile = Bp + ((blockIdx.y * nk + kt) << 11);
      async_b128(btile + tid * 8, sB + tid * 8);
    }

    // ---- stage A tile (128x32) ----
    if (CONV) {
      // implicit im2col gather (scalar, padding-predicated)
#pragma unroll
      for (int i = 0; i < 16; ++i) {
        int li = i * 256 + tid;
        int mi = li >> 5, kk = li & 31;
        int m = m0 + mi, k = k0 + kk;
        int img = m >> lHW;
        int rem = m & ((1 << lHW) - 1);
        int oh  = rem >> lOW;
        int ow  = rem & ((1 << lOW) - 1);
        int ci  = k >> 4;            // K = Ci*16 (4x4 kernel)
        int r   = k & 15;
        int kh  = r >> 2, kw = r & 3;
        int ih  = oh * 2 - 1 + kh;   // stride 2, pad 1
        int iw  = ow * 2 - 1 + kw;
        bool ok = (k < K) & ((unsigned)ih < (unsigned)IH) &
                  ((unsigned)iw < (unsigned)IW);
        u16 v = 0;
        if (ok) v = A[((img * Ci + ci) * IH + ih) * IW + iw];
        sA[mi * 32 + kk] = v;
      }
    } else {
      // contiguous rows: 2 x 16B vector moves per thread (lda, K mult. of 8)
#pragma unroll
      for (int i = 0; i < 2; ++i) {
        int li8 = i * 256 + tid;          // 8-half vector index
        int mi  = li8 >> 2, kk8 = (li8 & 3) * 8;
        int m = m0 + mi, k = k0 + kk8;
        u16x8 v = {0, 0, 0, 0, 0, 0, 0, 0};
        if ((m < M) & (k < K)) v = *(const u16x8*)&A[m * lda + k];
        *(u16x8*)&sA[mi * 32 + kk8] = v;
      }
    }

    wait_async0();       // wave-local async-to-LDS completion
    __syncthreads();     // cross-wave visibility of sA/sB

    // ---- gather WMMA fragments per ISA 7.12.2 bf16 layouts ----
    const int kbA = (lane >> 4) * 8;   // A: halves 0-7 -> K kbA..+7, 8-15 -> +16
    const int kbB = (lane >> 4) * 8;   // B: halves 0-7 -> K kbB..+7, 8-15 -> +16
    FragU afr[2], bfr[2];
#pragma unroll
    for (int ti = 0; ti < 2; ++ti) {
      int row = wm + ti * 16 + (lane & 15);
      afr[ti].h[0] = *(const u16x8*)&sA[row * 32 + kbA];
      afr[ti].h[1] = *(const u16x8*)&sA[row * 32 + kbA + 16];
    }
#pragma unroll
    for (int tj = 0; tj < 2; ++tj) {
      int col = wn + tj * 16 + (lane & 15);
      bfr[tj].h[0] = *(const u16x8*)&sB[col * 32 + kbB];
      bfr[tj].h[1] = *(const u16x8*)&sB[col * 32 + kbB + 16];
    }

#pragma unroll
    for (int ti = 0; ti < 2; ++ti)
#pragma unroll
      for (int tj = 0; tj < 2; ++tj)
        acc[ti][tj] = __builtin_amdgcn_wmma_f32_16x16x32_bf16(
            false, afr[ti].bf, false, bfr[tj].bf, (short)0, acc[ti][tj],
            false, false);
    __syncthreads();
  }

  // ---- epilogue: bias add, scatter (conv: back to NCHW) ----
  const int n0 = blockIdx.y * 64;
#pragma unroll
  for (int ti = 0; ti < 2; ++ti) {
#pragma unroll
    for (int tj = 0; tj < 2; ++tj) {
      int ncol = n0 + wn + tj * 16 + (lane & 15);
      if (ncol >= N) continue;
      float bv = bias ? bias[ncol] : 0.f;
#pragma unroll
      for (int v = 0; v < 8; ++v) {
        int m = m0 + wm + ti * 16 + (lane >> 4) * 8 + v;  // C/D: M = v + 8*(lane>=16)
        if (!CONV && m >= M) continue;
        float val = acc[ti][tj][v] + bv;
        if (CONV) {
          int img = m >> lHW;
          int rem = m & ((1 << lHW) - 1);
          int oh  = rem >> lOW;
          int ow  = rem & ((1 << lOW) - 1);
          C[((img * N + ncol) << lHW) + (oh << lOW) + ow] = val;
        } else {
          C[m * ldc + ncol] = val;
        }
      }
    }
  }
}

// ---------------------------------------------------------------------------
// Per-sample LayerNorm over (C,H,W) + SiLU, output bf16. 1 block per sample.
// ---------------------------------------------------------------------------
__global__ __launch_bounds__(256) void ln_silu_kernel(
    const float* __restrict__ in, const float* __restrict__ g,
    const float* __restrict__ b, u16* __restrict__ out, int chw) {
  __shared__ float rs[256], rq[256];
  const int s   = blockIdx.x;
  const int tid = threadIdx.x;
  const float* x = in + s * chw;
  float sum = 0.f, sq = 0.f;
  for (int i = tid; i < chw; i += 256) {
    float v = x[i];
    sum += v;
    sq  += v * v;
  }
  rs[tid] = sum;
  rq[tid] = sq;
  __syncthreads();
  for (int off = 128; off > 0; off >>= 1) {
    if (tid < off) { rs[tid] += rs[tid + off]; rq[tid] += rq[tid + off]; }
    __syncthreads();
  }
  float mean = rs[0] / (float)chw;
  float var  = rq[0] / (float)chw - mean * mean;
  float inv  = rsqrtf(var + 1e-5f);
  u16* o = out + s * chw;
  for (int i = tid; i < chw; i += 256) {
    float y  = (x[i] - mean) * inv * g[i] + b[i];
    float sl = y / (1.f + expf(-y));   // SiLU
    o[i] = f2bfu(sl);
  }
}

// ---------------------------------------------------------------------------
// GRU step part 1: gates + h_new. One block per batch row b, 128 threads (j).
// z is one-hot -> x@W_ih^T collapses to 32 gathered W_ih columns per gate.
// ---------------------------------------------------------------------------
__global__ __launch_bounds__(128) void gru_step1_kernel(
    const float* __restrict__ hprev, float* __restrict__ hnext,
    const float* __restrict__ actions,   // [B,T,6]
    const float* __restrict__ feats,     // [B*T,256]
    const float* __restrict__ wih,       // [384,1030]
    const float* __restrict__ whh,       // [384,128]
    const float* __restrict__ bih, const float* __restrict__ bhh,
    const int* __restrict__ zidx,        // [16,32] argmax indices (prev step)
    u16* __restrict__ xpost,             // [16,384] bf16 [h_new | f_t]
    float* __restrict__ hout,            // d_out h region
    int t, int first) {
  __shared__ float hs[128];
  __shared__ float as[8];
  const int b = blockIdx.x;
  const int j = threadIdx.x;
  hs[j] = first ? 0.f : hprev[b * 128 + j];
  if (j < 6) as[j] = actions[(b * 64 + t) * 6 + j];
  __syncthreads();

  float gi[3], gh[3];
#pragma unroll
  for (int g = 0; g < 3; ++g) {
    int n = g * 128 + j;
    const float* wr = wih + n * 1030;
    float a = bih[n];
    if (!first) {
#pragma unroll
      for (int nc = 0; nc < 32; ++nc) a += wr[nc * 32 + zidx[b * 32 + nc]];
    }
#pragma unroll
    for (int k = 0; k < 6; ++k) a += as[k] * wr[1024 + k];
    gi[g] = a;
    const float* hr = whh + n * 128;
    float c = bhh[n];
    for (int k = 0; k < 128; ++k) c += hs[k] * hr[k];
    gh[g] = c;
  }
  float r  = 1.f / (1.f + expf(-(gi[0] + gh[0])));
  float u  = 1.f / (1.f + expf(-(gi[1] + gh[1])));
  float nn = tanhf(gi[2] + r * gh[2]);
  float hn = (1.f - u) * nn + u * hs[j];

  hnext[b * 128 + j]           = hn;
  hout[(b * 64 + t) * 128 + j] = hn;
  xpost[b * 384 + j]           = f2bfu(hn);
  for (int i = j; i < 256; i += 128)
    xpost[b * 384 + 128 + i] = f2bfu(feats[(b * 64 + t) * 256 + i]);
}

// ---------------------------------------------------------------------------
// GRU step part 3: per-(b,nc) argmax over 32 classes -> one-hot z + indices
// ---------------------------------------------------------------------------
__global__ void gru_argmax_kernel(const float* __restrict__ post,
                                  float* __restrict__ zout,
                                  int* __restrict__ zidx, int t) {
  const int b  = blockIdx.x;
  const int nc = threadIdx.x;  // 32
  const float* p = post + (b * 64 + t) * 1024 + nc * 32;
  int best = 0;
  float bv = p[0];
#pragma unroll
  for (int cs = 1; cs < 32; ++cs) {
    float v = p[cs];
    if (v > bv) { bv = v; best = cs; }
  }
  zidx[b * 32 + nc] = best;
  float* zo = zout + (b * 64 + t) * 1024 + nc * 32;
#pragma unroll
  for (int cs = 0; cs < 32; ++cs) zo[cs] = (cs == best) ? 1.f : 0.f;
}

// ---------------------------------------------------------------------------
// Host launcher
// ---------------------------------------------------------------------------
extern "C" void kernel_launch(void* const* d_in, const int* in_sizes, int n_in,
                              void* d_out, int out_size, void* d_ws, size_t ws_size,
                              hipStream_t stream) {
  (void)in_sizes; (void)n_in; (void)out_size; (void)ws_size;
  const float* states  = (const float*)d_in[0];
  const float* actions = (const float*)d_in[1];
  const float* c1w = (const float*)d_in[2];  const float* c1b = (const float*)d_in[3];
  const float* l1g = (const float*)d_in[4];  const float* l1b = (const float*)d_in[5];
  const float* c2w = (const float*)d_in[6];  const float* c2b = (const float*)d_in[7];
  const float* l2g = (const float*)d_in[8];  const float* l2b = (const float*)d_in[9];
  const float* c3w = (const float*)d_in[10]; const float* c3b = (const float*)d_in[11];
  const float* l3g = (const float*)d_in[12]; const float* l3b = (const float*)d_in[13];
  const float* c4w = (const float*)d_in[14]; const float* c4b = (const float*)d_in[15];
  const float* l4g = (const float*)d_in[16]; const float* l4b = (const float*)d_in[17];
  const float* fcw = (const float*)d_in[18]; const float* fcb = (const float*)d_in[19];
  const float* wih = (const float*)d_in[20]; const float* whh = (const float*)d_in[21];
  const float* bih = (const float*)d_in[22]; const float* bhh = (const float*)d_in[23];
  const float* prw = (const float*)d_in[24]; const float* prb = (const float*)d_in[25];
  const float* pow_ = (const float*)d_in[26]; const float* pob = (const float*)d_in[27];

  // Workspace carve-up
  char* wp = (char*)d_ws;
  auto alloc = [&](size_t bytes) -> void* {
    void* p = (void*)wp;
    wp += (bytes + 255) & ~(size_t)255;
    return p;
  };
  u16*   x0   = (u16*)alloc(12582912ull * 2);   // states bf16 [1024,3,64,64]
  float* cbuf = (float*)alloc(33554432ull * 4); // conv out scratch (reused L1..L4)
  u16*   a1   = (u16*)alloc(33554432ull * 2);   // [1024,32,32,32] bf16
  u16*   a2   = (u16*)alloc(16777216ull * 2);
  u16*   a3   = (u16*)alloc(8388608ull * 2);
  u16*   a4   = (u16*)alloc(4194304ull * 2);
  float* feats = (float*)alloc(262144ull * 4);  // [1024,256]
  // tile-packed weights: Nb*NK blocks of 64x32 bf16 (4 KB each)
  u16* w1p = (u16*)alloc((size_t)1 * 2 * 2048 * 2);
  u16* w2p = (u16*)alloc((size_t)1 * 16 * 2048 * 2);
  u16* w3p = (u16*)alloc((size_t)2 * 32 * 2048 * 2);
  u16* w4p = (u16*)alloc((size_t)4 * 64 * 2048 * 2);
  u16* fcp = (u16*)alloc((size_t)4 * 128 * 2048 * 2);
  u16* prp = (u16*)alloc((size_t)16 * 4 * 2048 * 2);
  u16* pop = (u16*)alloc((size_t)16 * 12 * 2048 * 2);
  float* hA    = (float*)alloc(16 * 128 * 4);
  float* hB    = (float*)alloc(16 * 128 * 4);
  u16*   xpost = (u16*)alloc(16 * 384 * 2);
  int*   zidx  = (int*)alloc(16 * 32 * 4);

  float* out      = (float*)d_out;
  float* prior_o  = out;                 // [B,T,32,32]
  float* post_o   = out + 1048576;       // [B,T,32,32]
  float* h_o      = out + 2097152;       // [B,T,128]
  float* z_o      = out + 2228224;       // [B,T,1024]

  // init + packing
  zero_f32_kernel<<<8, 256, 0, stream>>>(hA, 2048);
  cvt_f32_bf16_kernel<<<(12582912 + 255) / 256, 256, 0, stream>>>(states, x0, 12582912);
  pack_wTtile_kernel<<<(1 * 2 * 2048 + 255) / 256, 256, 0, stream>>>(c1w, w1p, 32, 48, 2);
  pack_wTtile_kernel<<<(1 * 16 * 2048 + 255) / 256, 256, 0, stream>>>(c2w, w2p, 64, 512, 16);
  pack_wTtile_kernel<<<(2 * 32 * 2048 + 255) / 256, 256, 0, stream>>>(c3w, w3p, 128, 1024, 32);
  pack_wTtile_kernel<<<(4 * 64 * 2048 + 255) / 256, 256, 0, stream>>>(c4w, w4p, 256, 2048, 64);
  pack_wTtile_kernel<<<(4 * 128 * 2048 + 255) / 256, 256, 0, stream>>>(fcw, fcp, 256, 4096, 128);
  pack_wTtile_kernel<<<(16 * 4 * 2048 + 255) / 256, 256, 0, stream>>>(prw, prp, 1024, 128, 4);
  pack_wTtile_kernel<<<(16 * 12 * 2048 + 255) / 256, 256, 0, stream>>>(pow_, pop, 1024, 384, 12);

  // Encoder: conv(im2col-WMMA) -> LN+SiLU, x4
  gemm_wmma_kernel<1><<<dim3(8192, 1), 256, 0, stream>>>(
      x0, w1p, c1b, cbuf, 1048576, 32, 48, 0, 0, 3, 64, 64, 32);
  ln_silu_kernel<<<1024, 256, 0, stream>>>(cbuf, l1g, l1b, a1, 32768);

  gemm_wmma_kernel<1><<<dim3(2048, 1), 256, 0, stream>>>(
      a1, w2p, c2b, cbuf, 262144, 64, 512, 0, 0, 32, 32, 32, 16);
  ln_silu_kernel<<<1024, 256, 0, stream>>>(cbuf, l2g, l2b, a2, 16384);

  gemm_wmma_kernel<1><<<dim3(512, 2), 256, 0, stream>>>(
      a2, w3p, c3b, cbuf, 65536, 128, 1024, 0, 0, 64, 16, 16, 8);
  ln_silu_kernel<<<1024, 256, 0, stream>>>(cbuf, l3g, l3b, a3, 8192);

  gemm_wmma_kernel<1><<<dim3(128, 4), 256, 0, stream>>>(
      a3, w4p, c4b, cbuf, 16384, 256, 2048, 0, 0, 128, 8, 8, 4);
  ln_silu_kernel<<<1024, 256, 0, stream>>>(cbuf, l4g, l4b, a4, 4096);

  // FC: [1024,4096] x [4096,256]
  gemm_wmma_kernel<0><<<dim3(8, 4), 256, 0, stream>>>(
      a4, fcp, fcb, feats, 1024, 256, 4096, 4096, 256, 0, 0, 0, 4);

  // Sequential GRU: 64 steps, WMMA heads (prior 16x1024x128, post 16x1024x384)
  float* hprev = hA;
  float* hnext = hB;
  for (int t = 0; t < 64; ++t) {
    gru_step1_kernel<<<16, 128, 0, stream>>>(
        hprev, hnext, actions, feats, wih, whh, bih, bhh, zidx, xpost, h_o, t,
        (t == 0) ? 1 : 0);
    gemm_wmma_kernel<0><<<dim3(1, 16), 256, 0, stream>>>(
        xpost, prp, prb, prior_o + t * 1024, 16, 1024, 128, 384, 65536,
        0, 0, 0, 4);
    gemm_wmma_kernel<0><<<dim3(1, 16), 256, 0, stream>>>(
        xpost, pop, pob, post_o + t * 1024, 16, 1024, 384, 384, 65536,
        0, 0, 0, 4);
    gru_argmax_kernel<<<16, 32, 0, stream>>>(post_o, z_o, zidx, t);
    float* tmp = hprev; hprev = hnext; hnext = tmp;
  }
}
